// Tree_net_39694087750206
// MI455X (gfx1250) — matrix operand order repaired and checked by reading, not compiled, not run on hardware
//
#include <hip/hip_runtime.h>
#include <hip/hip_bf16.h>

typedef __attribute__((ext_vector_type(16))) _Float16 v16h;
typedef __attribute__((ext_vector_type(8)))  float    v8f;
typedef __attribute__((ext_vector_type(4)))  int      v4i;

#define D_DIM 2048
#define N_DIM 2047
#define NPAD  2048
#define M_DIM 2048
#define C_DIM 8
#define K2    (2 * NPAD)   // 4096: fused [logsig(z) | z] K dimension

#define AS1 __attribute__((address_space(1)))
#define AS3 __attribute__((address_space(3)))

#if __has_builtin(__builtin_amdgcn_global_load_async_to_lds_b128) && \
    __has_builtin(__builtin_amdgcn_s_wait_asynccnt)
#define HAS_ASYNC_LDS 1
#else
#define HAS_ASYNC_LDS 0
#endif

// ---------------------------------------------------------------------------
// Kernel 0: out[b][c] = b2[c]  (bias init; gemm2 accumulates on top)
// ---------------------------------------------------------------------------
__global__ __launch_bounds__(256) void init_out_kernel(float* __restrict__ out,
                                                       const float* __restrict__ b2,
                                                       int total) {
  int i = blockIdx.x * 256 + threadIdx.x;
  if (i < total) out[i] = b2[i & (C_DIM - 1)];
}

// ---------------------------------------------------------------------------
// Kernel 1: z = 10*(inp @ W1^T + b1); A2[b][n] = (f16)logsig(z), A2[b][NPAD+n] = (f16)z
// 256 threads = 8 waves in a 4x2 grid; each wave owns a 2x4 grid of 16x16 C
// tiles (32 rows x 64 cols). Block tile: 128 rows x 128 cols.
// grid = (NPAD/128, B/128).
// ---------------------------------------------------------------------------
__global__ __launch_bounds__(256) void gemm1_kernel(const float* __restrict__ inp,
                                                    const float* __restrict__ W1,
                                                    const float* __restrict__ b1,
                                                    _Float16* __restrict__ A2) {
  __shared__ _Float16 As[128][40];   // [row][k]
  __shared__ _Float16 Bs[128][40];   // [col(n)][k] (transposed: k-pairs contiguous)

  const int tid  = threadIdx.x;
  const int wave = tid >> 5;
  const int lane = tid & 31;
  const int wr = wave >> 1;          // 0..3 (row group of 32)
  const int wc = wave & 1;           // 0..1 (col group of 64)
  const int lr = lane & 15;
  const int hi = lane >> 4;

  const int  n0 = blockIdx.x * 128;
  const long r0 = (long)blockIdx.y * 128;

  // register staging for pipelining
  const int ar   = tid >> 1;               // 0..127
  const int akof = (tid & 1) * 16;         // 0 or 16
  const int bc   = tid >> 1;               // 0..127 (column n in tile)
  const int bkof = (tid & 1) * 16;         // 0 or 16

  float4 areg[4];
  float4 breg[4];

  auto load_tiles = [&](int k0) {
    const float4* ap = reinterpret_cast<const float4*>(inp + (r0 + ar) * D_DIM + k0 + akof);
    areg[0] = ap[0]; areg[1] = ap[1]; areg[2] = ap[2]; areg[3] = ap[3];
    breg[0] = breg[1] = breg[2] = breg[3] = float4{0.f, 0.f, 0.f, 0.f};
    if (n0 + bc < N_DIM) {
      const float4* bp = reinterpret_cast<const float4*>(W1 + (long)(n0 + bc) * D_DIM + k0 + bkof);
      breg[0] = bp[0]; breg[1] = bp[1]; breg[2] = bp[2]; breg[3] = bp[3];
      __builtin_prefetch(W1 + (long)(n0 + bc) * D_DIM + k0 + bkof + 32, 0, 1);
    }
  };
  auto store_tiles = [&]() {
    _Float16* ad = &As[ar][akof];
#pragma unroll
    for (int q = 0; q < 4; ++q) {
      ad[4 * q + 0] = (_Float16)areg[q].x; ad[4 * q + 1] = (_Float16)areg[q].y;
      ad[4 * q + 2] = (_Float16)areg[q].z; ad[4 * q + 3] = (_Float16)areg[q].w;
    }
    _Float16* bd = &Bs[bc][bkof];
#pragma unroll
    for (int q = 0; q < 4; ++q) {
      bd[4 * q + 0] = (_Float16)breg[q].x; bd[4 * q + 1] = (_Float16)breg[q].y;
      bd[4 * q + 2] = (_Float16)breg[q].z; bd[4 * q + 3] = (_Float16)breg[q].w;
    }
  };

  v8f acc[2][4] = {};
  load_tiles(0);

  for (int k0 = 0; k0 < D_DIM; k0 += 32) {
    __syncthreads();                 // previous compute done reading LDS
    store_tiles();
    if (k0 + 32 < D_DIM) load_tiles(k0 + 32);   // overlap next loads with WMMA
    __syncthreads();                 // tiles visible

    v16h af[2], bf[4];
#pragma unroll
    for (int t = 0; t < 2; ++t) {
      const _Float16* arow = &As[wr * 32 + t * 16 + lr][0];
#pragma unroll
      for (int p = 0; p < 8; ++p) {
        int ka = ((p < 4) ? (2 * p) : (8 + 2 * p)) + hi * 8;
        af[t][2 * p] = arow[ka]; af[t][2 * p + 1] = arow[ka + 1];
      }
    }
#pragma unroll
    for (int t = 0; t < 4; ++t) {
      const _Float16* bcol = &Bs[wc * 64 + t * 16 + lr][0];
#pragma unroll
      for (int p = 0; p < 8; ++p) {
        int kb = ((p < 4) ? (2 * p) : (8 + 2 * p)) + hi * 8;
        bf[t][2 * p] = bcol[kb]; bf[t][2 * p + 1] = bcol[kb + 1];
      }
    }
#pragma unroll
    for (int tr = 0; tr < 2; ++tr)
#pragma unroll
      for (int tc = 0; tc < 4; ++tc)
        acc[tr][tc] = __builtin_amdgcn_wmma_f32_16x16x32_f16(
            false, af[tr], false, bf[tc], (short)0, acc[tr][tc], false, false);
  }

  // --- epilogue: z, logsig(z) -> fused f16 A-operand for stage 2 ---
#pragma unroll
  for (int tc = 0; tc < 4; ++tc) {
    const int n = n0 + wc * 64 + tc * 16 + lr;
    const float bias = (n < N_DIM) ? b1[n] : 0.f;
#pragma unroll
    for (int tr = 0; tr < 2; ++tr) {
#pragma unroll
      for (int i = 0; i < 8; ++i) {
        int row_l = wr * 32 + tr * 16 + hi * 8 + i;
        long b = r0 + row_l;
        float z = 0.f, ls = 0.f;
        if (n < N_DIM) {
          z  = 10.f * (acc[tr][tc][i] + bias);
          ls = fminf(z, 0.f) - log1pf(__expf(-fabsf(z)));  // stable log_sigmoid
        }
        A2[b * K2 + n]        = (_Float16)ls;
        A2[b * K2 + NPAD + n] = (_Float16)z;
      }
    }
  }
}

// ---------------------------------------------------------------------------
// Kernel 2: log_xc = A2 @ U  (U synthesized from b_mat: [|b|; -(b==1)]),
//           xc = exp(log_xc), out += xc @ W2^T  (LDS-reduced, then atomics)
// Same 128x128 block tiling. grid = (M/128, B/128).
// A-tile staged via GLOBAL_LOAD_ASYNC_TO_LDS_B128 when available.
// ---------------------------------------------------------------------------
__global__ __launch_bounds__(256) void gemm2_kernel(const _Float16* __restrict__ A2,
                                                    const int* __restrict__ bmat,
                                                    const float* __restrict__ W2,
                                                    float* __restrict__ out) {
  __shared__ _Float16 As[128][40];
  __shared__ _Float16 Bs[128][40];
  __shared__ float    redc[128][C_DIM];

  const int tid  = threadIdx.x;
  const int wave = tid >> 5;
  const int lane = tid & 31;
  const int wr = wave >> 1;
  const int wc = wave & 1;
  const int lr = lane & 15;
  const int hi = lane >> 4;

  const int  m0 = blockIdx.x * 128;
  const long r0 = (long)blockIdx.y * 128;

  for (int idx = tid; idx < 128 * C_DIM; idx += 256)
    (&redc[0][0])[idx] = 0.f;

  const int ar   = tid >> 1;            // 0..127
  const int akof = (tid & 1) * 16;      // 0 or 16 (f16 elements)
  const int bc   = tid >> 1;            // 0..127 (column m in tile)
  const int bj0  = (tid & 1) * 16;      // 0 or 16

  v8f acc[2][4] = {};

  for (int k0 = 0; k0 < K2; k0 += 32) {
    __syncthreads();                    // previous compute done reading LDS

    // --- A tile: 128x32 f16 straight copy global -> LDS ---
    const _Float16* gp = A2 + (r0 + ar) * K2 + k0 + akof;
    _Float16* lp = &As[ar][akof];
#if HAS_ASYNC_LDS
    __builtin_amdgcn_global_load_async_to_lds_b128((AS1 v4i*)gp,       (AS3 v4i*)lp,       0, 0);
    __builtin_amdgcn_global_load_async_to_lds_b128((AS1 v4i*)(gp + 8), (AS3 v4i*)(lp + 8), 0, 0);
#else
    const uint4* gv = reinterpret_cast<const uint4*>(gp);
    reinterpret_cast<uint4*>(lp)[0] = gv[0];
    reinterpret_cast<uint4*>(lp)[1] = gv[1];
#endif
    if (k0 + 32 < K2)
      __builtin_prefetch(A2 + (r0 + ar) * K2 + k0 + 32 + akof, 0, 1);

    // --- B tile: synthesize U from b_mat: k<NPAD -> |b|, k>=NPAD -> -(b==1) ---
    {
      long m = m0 + bc;
#pragma unroll
      for (int j = 0; j < 16; ++j) {
        int kk = k0 + bj0 + j;
        float u = 0.f;
        if (kk < N_DIM) {
          u = (bmat[m * N_DIM + kk] != 0) ? 1.f : 0.f;
        } else if (kk >= NPAD) {
          int k = kk - NPAD;
          if (k < N_DIM) u = (bmat[m * N_DIM + k] == 1) ? -1.f : 0.f;
        }
        Bs[bc][bj0 + j] = (_Float16)u;
      }
    }

#if HAS_ASYNC_LDS
    __builtin_amdgcn_s_wait_asynccnt(0);   // A tile landed in LDS
#endif
    __syncthreads();                       // publish tiles

    v16h af[2], bf[4];
#pragma unroll
    for (int t = 0; t < 2; ++t) {
      const _Float16* arow = &As[wr * 32 + t * 16 + lr][0];
#pragma unroll
      for (int p = 0; p < 8; ++p) {
        int ka = ((p < 4) ? (2 * p) : (8 + 2 * p)) + hi * 8;
        af[t][2 * p] = arow[ka]; af[t][2 * p + 1] = arow[ka + 1];
      }
    }
#pragma unroll
    for (int t = 0; t < 4; ++t) {
      const _Float16* bcol = &Bs[wc * 64 + t * 16 + lr][0];
#pragma unroll
      for (int p = 0; p < 8; ++p) {
        int kb = ((p < 4) ? (2 * p) : (8 + 2 * p)) + hi * 8;
        bf[t][2 * p] = bcol[kb]; bf[t][2 * p + 1] = bcol[kb + 1];
      }
    }
#pragma unroll
    for (int tr = 0; tr < 2; ++tr)
#pragma unroll
      for (int tc = 0; tc < 4; ++tc)
        acc[tr][tc] = __builtin_amdgcn_wmma_f32_16x16x32_f16(
            false, af[tr], false, bf[tc], (short)0, acc[tr][tc], false, false);
  }

  // --- epilogue: xc = exp(log_xc); project onto W2 (C=8) via LDS reduction ---
#pragma unroll
  for (int tc = 0; tc < 4; ++tc) {
    const int m = m0 + wc * 64 + tc * 16 + lr;
    float w2v[C_DIM];
#pragma unroll
    for (int c = 0; c < C_DIM; ++c) w2v[c] = W2[c * M_DIM + m];
#pragma unroll
    for (int tr = 0; tr < 2; ++tr) {
#pragma unroll
      for (int i = 0; i < 8; ++i) {
        int row_l = wr * 32 + tr * 16 + hi * 8 + i;
        float xc = __expf(acc[tr][tc][i]);
#pragma unroll
        for (int c = 0; c < C_DIM; ++c)
          atomicAdd(&redc[row_l][c], xc * w2v[c]);
      }
    }
  }
  __syncthreads();

  for (int idx = tid; idx < 128 * C_DIM; idx += 256) {
    int row_l = idx >> 3, c = idx & 7;
    atomicAdd(&out[(r0 + row_l) * C_DIM + c], redc[row_l][c]);
  }
}

// ---------------------------------------------------------------------------
extern "C" void kernel_launch(void* const* d_in, const int* in_sizes, int n_in,
                              void* d_out, int out_size, void* d_ws, size_t ws_size,
                              hipStream_t stream) {
  const float* inp  = (const float*)d_in[0];
  const float* W1   = (const float*)d_in[1];
  const float* b1   = (const float*)d_in[2];
  const int*   bmat = (const int*)d_in[3];
  const float* W2   = (const float*)d_in[4];
  const float* b2   = (const float*)d_in[5];
  float* out = (float*)d_out;

  const int B = in_sizes[0] / D_DIM;          // 8192
  _Float16* A2 = (_Float16*)d_ws;             // [B][K2] f16 = B*4096*2 bytes

  dim3 blk(256);
  init_out_kernel<<<dim3((B * C_DIM + 255) / 256), blk, 0, stream>>>(out, b2, B * C_DIM);

  dim3 g1(NPAD / 128, B / 128);
  gemm1_kernel<<<g1, blk, 0, stream>>>(inp, W1, b1, A2);

  dim3 g2(M_DIM / 128, B / 128);
  gemm2_kernel<<<g2, blk, 0, stream>>>(A2, bmat, W2, out);
}
